// AttentionBasedScaleIntegrationUnit_11166914970124
// MI455X (gfx1250) — matrix operand order repaired and checked by reading, not compiled, not run on hardware
//
#include <hip/hip_runtime.h>
#include <hip/hip_bf16.h>

#define S_ 3
#define B_ 4
#define C_ 256
#define HW_ 4096
#define N_ 16384   // B*H*W
#define NT_ 4      // N-tiles of 16 per wave -> 64-wide position tile
#define TILE_N 64
#define EPS_ 1e-5f

typedef __attribute__((ext_vector_type(16))) _Float16 v16h;
typedef __attribute__((ext_vector_type(8)))  _Float16 v8h;
typedef __attribute__((ext_vector_type(8)))  float    v8f;

__device__ __forceinline__ float gelu_exact(float x) {
    return 0.5f * x * (1.0f + erff(x * 0.70710678118654752f));
}

__device__ __forceinline__ v16h cat8(v8h lo, v8h hi) {
    return __builtin_shufflevector(lo, hi, 0,1,2,3,4,5,6,7,8,9,10,11,12,13,14,15);
}

// ---- WMMA fragment loaders (ISA 7.12.2, wave32) ---------------------------
// A 16x32 f16 fragment straight from GLOBAL f16 weights (row-major, ld=lda).
// lane<16: row=lane, K kb..kb+7 then 16+kb.. ; kb=0 / 8 by lane half.
__device__ __forceinline__ v16h load_a_g(const _Float16* __restrict__ A, int lda,
                                         int cka, int m0, int lane) {
    int row = m0 + (lane & 15);
    int kb  = (lane < 16) ? 0 : 8;
    const _Float16* p = A + (size_t)row * lda + cka + kb;
    v8h lo = *(const v8h*)p;         // K kb..kb+7
    v8h hi = *(const v8h*)(p + 16);  // K 16+kb..16+kb+7
    return cat8(lo, hi);
}
// B 32x16 fragment from a TRANSPOSED LDS panel Bt[col][k] (k contiguous).
// lanes 0-15: K 0..7,16..23, lanes 16-31: K 8..15,24..31 -> two b128 loads.
__device__ __forceinline__ v16h load_bt(const _Float16* __restrict__ Bt, int ldk,
                                        int ckb, int n0, int lane) {
    int col = n0 + (lane & 15);
    int kb  = (lane < 16) ? 0 : 8;
    const _Float16* p = Bt + (size_t)col * ldk + ckb + kb;
    v8h lo = *(const v8h*)p;
    v8h hi = *(const v8h*)(p + 16);
    return cat8(lo, hi);
}
// One K=32 step: wave owns M-strip [m0,m0+32) x 64 columns -> 8 WMMAs.
__device__ __forceinline__ void gemm_step_g(const _Float16* __restrict__ Ag, int lda, int cka,
                                            const _Float16* __restrict__ Bt, int ldk, int ckb,
                                            int m0, int lane, v8f* acc /*[8]*/) {
    v16h a0 = load_a_g(Ag, lda, cka, m0, lane);
    v16h a1 = load_a_g(Ag, lda, cka, m0 + 16, lane);
#pragma unroll
    for (int nt = 0; nt < NT_; ++nt) {
        v16h bf = load_bt(Bt, ldk, ckb, nt * 16, lane);
        acc[nt] = __builtin_amdgcn_wmma_f32_16x16x32_f16(
            false, a0, false, bf, (short)0, acc[nt], false, false);
        acc[NT_ + nt] = __builtin_amdgcn_wmma_f32_16x16x32_f16(
            false, a1, false, bf, (short)0, acc[NT_ + nt], false, false);
    }
}

// ===========================================================================
// Weight pre-conversion: f32 -> f16 (done once; weights then live in L2)
// ===========================================================================
__global__ __launch_bounds__(256) void cvt_lin_kernel(const float* __restrict__ src,
                                                      _Float16* __restrict__ dst, int n) {
    int i = blockIdx.x * 256 + threadIdx.x;
    if (i < n) dst[i] = (_Float16)src[i];
}
// op_w1 (O,C,3,3) -> w1h[tap][o][c]
__global__ __launch_bounds__(256) void cvt_conv_kernel(const float* __restrict__ w1,
                                                       _Float16* __restrict__ w1h) {
    int i = blockIdx.x * 256 + threadIdx.x;       // 9*65536
    int tap = i >> 16, rem = i & 65535, o = rem >> 8, c = rem & 255;
    w1h[i] = (_Float16)w1[(size_t)(o * 256 + c) * 9 + tap];
}

// ===========================================================================
// Kernel P: pooled[b, s*C+c] = mean_hw x[s,b,c,:,:]
// ===========================================================================
__global__ __launch_bounds__(256) void pool_kernel(const float* __restrict__ x,
                                                   float* __restrict__ pooled) {
    __shared__ float red[256];
    int c = blockIdx.x & 255;
    int b = (blockIdx.x >> 8) & 3;
    int s = blockIdx.x >> 10;
    const float4* xv = (const float4*)(x + ((size_t)(s * B_ + b) * C_ + c) * HW_);
    float acc = 0.f;
    for (int i = threadIdx.x; i < HW_ / 4; i += 256) {
        float4 v = xv[i];
        acc += v.x + v.y + v.z + v.w;
    }
    red[threadIdx.x] = acc;
    __syncthreads();
    for (int off = 128; off > 0; off >>= 1) {
        if (threadIdx.x < off) red[threadIdx.x] += red[threadIdx.x + off];
        __syncthreads();
    }
    if (threadIdx.x == 0) pooled[b * (S_ * C_) + s * C_ + c] = red[0] * (1.0f / HW_);
}

// ===========================================================================
// Kernel A: importance MLP + softmax -> wgt[b][s]
// ===========================================================================
__global__ __launch_bounds__(256) void mlp_kernel(const float* __restrict__ pooled,
                                                  const float* __restrict__ w1,
                                                  const float* __restrict__ b1,
                                                  const float* __restrict__ w2,
                                                  const float* __restrict__ b2,
                                                  float* __restrict__ wgt) {
    __shared__ float hL[B_ * C_];
    __shared__ float lg[B_ * S_];
    int j = threadIdx.x;
    for (int b = 0; b < B_; ++b) {
        float acc = b1[j];
        for (int i = 0; i < S_ * C_; ++i) acc += pooled[b * (S_ * C_) + i] * w1[j * (S_ * C_) + i];
        hL[b * C_ + j] = gelu_exact(acc);
    }
    __syncthreads();
    if (j < B_ * S_) {
        int b = j / S_, s = j % S_;
        float acc = b2[s];
        for (int i = 0; i < C_; ++i) acc += hL[b * C_ + i] * w2[s * C_ + i];
        lg[j] = acc;
    }
    __syncthreads();
    if (j < B_) {
        float l0 = lg[j * S_], l1 = lg[j * S_ + 1], l2 = lg[j * S_ + 2];
        float mx = fmaxf(l0, fmaxf(l1, l2));
        float e0 = expf(l0 - mx), e1 = expf(l1 - mx), e2 = expf(l2 - mx);
        float inv = 1.0f / (e0 + e1 + e2);
        wgt[j * S_ + 0] = e0 * inv; wgt[j * S_ + 1] = e1 * inv; wgt[j * S_ + 2] = e2 * inv;
    }
}

// ===========================================================================
// Kernel B: fused  y = gelu(BN(W_t x + b_t)) ; qkv = Wi_t y + bi_t
// qkv layout: [t][s][n][c] (f16, c contiguous)
// ===========================================================================
__global__ __launch_bounds__(256) void proj_kernel(
    const float* __restrict__ x,
    const _Float16* __restrict__ Wqh, const _Float16* __restrict__ Wkh,
    const _Float16* __restrict__ Wvh,
    const float* __restrict__ bq, const float* __restrict__ bnq,
    const float* __restrict__ bk, const float* __restrict__ bnk,
    const float* __restrict__ bv, const float* __restrict__ bnv,
    const _Float16* __restrict__ aiwh, const float* __restrict__ aib,
    _Float16* __restrict__ qkv) {
    __shared__ _Float16 XpT[TILE_N * 32];   // [p][k]  4 KB
    __shared__ _Float16 YpT[TILE_N * C_];   // [p][c] 32 KB
    int n0 = blockIdx.x * TILE_N;
    int s  = blockIdx.y;
    int b  = n0 >> 12, hw0 = n0 & (HW_ - 1);
    int tid = threadIdx.x, lane = tid & 31, m0 = (tid >> 5) * 32;
    int sp = tid & 63, k0 = (tid >> 6) * 8;   // staging coords

    for (int t = 0; t < 3; ++t) {
        const _Float16* Wh = (t == 0) ? Wqh : (t == 1) ? Wkh : Wvh;
        const float* bb = (t == 0) ? bq : (t == 1) ? bk : bv;
        const float* bn = (t == 0) ? bnq : (t == 1) ? bnk : bnv;

        // ---- GEMM1: Wh[s] (256x256) x X (256x64) ----
        v8f acc[2 * NT_];
#pragma unroll
        for (int i = 0; i < 2 * NT_; ++i) acc[i] = (v8f){};
        for (int ck = 0; ck < C_; ck += 32) {
            __syncthreads();
            v8h hx;                                   // transpose-stage x -> XpT[p][k]
#pragma unroll
            for (int j = 0; j < 8; ++j)
                hx[j] = (_Float16)x[((size_t)(s * B_ + b) * C_ + ck + k0 + j) * HW_ + hw0 + sp];
            *(v8h*)(XpT + sp * 32 + k0) = hx;
            if (ck + 32 < C_)
                __builtin_prefetch(&x[((size_t)(s * B_ + b) * C_ + ck + 32 + k0) * HW_ + hw0 + sp], 0, 0);
            __syncthreads();
            gemm_step_g(Wh + (size_t)s * C_ * C_, C_, ck, XpT, 32, 0, m0, lane, acc);
        }
        // ---- epilogue1: bias + BN + GELU -> YpT[p][c] (one v8h store each) ----
        __syncthreads();
#pragma unroll
        for (int mt = 0; mt < 2; ++mt)
#pragma unroll
            for (int nt = 0; nt < NT_; ++nt) {
                int p  = nt * 16 + (lane & 15);
                int rb = m0 + mt * 16 + ((lane >> 4) << 3);
                v8h hy;
#pragma unroll
                for (int v = 0; v < 8; ++v) {
                    int o = rb + v;
                    float g  = bn[(s * 4 + 0) * C_ + o];
                    float be = bn[(s * 4 + 1) * C_ + o];
                    float mu = bn[(s * 4 + 2) * C_ + o];
                    float va = bn[(s * 4 + 3) * C_ + o];
                    float y = acc[mt * NT_ + nt][v] + bb[s * C_ + o];
                    y = (y - mu) * (g * rsqrtf(va + EPS_)) + be;
                    hy[v] = (_Float16)gelu_exact(y);
                }
                *(v8h*)(YpT + p * C_ + rb) = hy;
            }
        __syncthreads();
        // ---- GEMM2: global A (aiwh row t*C..) x resident YpT -- no staging ----
        v8f acc2[2 * NT_];
#pragma unroll
        for (int i = 0; i < 2 * NT_; ++i) acc2[i] = (v8f){};
#pragma unroll
        for (int ck = 0; ck < C_; ck += 32)
            gemm_step_g(aiwh + (size_t)t * C_ * C_, C_, ck, YpT, C_, ck, m0, lane, acc2);
        // ---- epilogue2: + bias, v8h store to qkv[t][s][n][c] ----
#pragma unroll
        for (int mt = 0; mt < 2; ++mt)
#pragma unroll
            for (int nt = 0; nt < NT_; ++nt) {
                int p  = nt * 16 + (lane & 15);
                int rb = m0 + mt * 16 + ((lane >> 4) << 3);
                v8h hv;
#pragma unroll
                for (int v = 0; v < 8; ++v)
                    hv[v] = (_Float16)(acc2[mt * NT_ + nt][v] + aib[t * C_ + rb + v]);
                *(v8h*)(qkv + ((size_t)(t * S_ + s) * N_ + n0 + p) * C_ + rb) = hv;
            }
        __syncthreads();
    }
}

// ===========================================================================
// Kernel C: tiny attention (S=3) + softmax-weighted combine over s.
// Mws[n][c] = sum_s wgt[b][s] * o[n][s][c]
// ===========================================================================
__global__ __launch_bounds__(256) void attn_small_kernel(const _Float16* __restrict__ qkv,
                                                         const float* __restrict__ wgt,
                                                         _Float16* __restrict__ Mws) {
    int n = blockIdx.x * 256 + threadIdx.x;
    int b = n >> 12;
    float wb[3] = {wgt[b * S_ + 0], wgt[b * S_ + 1], wgt[b * S_ + 2]};
    const float sc = 0.17677669529663687f;  // 1/sqrt(32)
    for (int h = 0; h < 8; ++h) {
        int cb = h * 32;
        float kv[3][32];
#pragma unroll
        for (int t = 0; t < 3; ++t) {
            const _Float16* kp = qkv + ((size_t)(1 * S_ + t) * N_ + n) * C_ + cb;
#pragma unroll
            for (int j = 0; j < 4; ++j) {
                v8h hv = *(const v8h*)(kp + j * 8);
#pragma unroll
                for (int e = 0; e < 8; ++e) kv[t][j * 8 + e] = (float)hv[e];
            }
        }
        float coef[3] = {0.f, 0.f, 0.f};
#pragma unroll
        for (int s = 0; s < 3; ++s) {
            float qd[32];
            const _Float16* qp = qkv + ((size_t)(0 * S_ + s) * N_ + n) * C_ + cb;
#pragma unroll
            for (int j = 0; j < 4; ++j) {
                v8h hv = *(const v8h*)(qp + j * 8);
#pragma unroll
                for (int e = 0; e < 8; ++e) qd[j * 8 + e] = (float)hv[e];
            }
            float lg[3];
#pragma unroll
            for (int t = 0; t < 3; ++t) {
                float a = 0.f;
#pragma unroll
                for (int d = 0; d < 32; ++d) a += qd[d] * kv[t][d];
                lg[t] = a * sc;
            }
            float mx = fmaxf(lg[0], fmaxf(lg[1], lg[2]));
            float e0 = expf(lg[0] - mx), e1 = expf(lg[1] - mx), e2 = expf(lg[2] - mx);
            float inv = wb[s] / (e0 + e1 + e2);
            coef[0] += e0 * inv; coef[1] += e1 * inv; coef[2] += e2 * inv;
        }
#pragma unroll
        for (int t = 0; t < 3; ++t) {
            const _Float16* vp = qkv + ((size_t)(2 * S_ + t) * N_ + n) * C_ + cb;
#pragma unroll
            for (int j = 0; j < 4; ++j) {
                v8h hv = *(const v8h*)(vp + j * 8);
#pragma unroll
                for (int e = 0; e < 8; ++e) kv[t][j * 8 + e] = (float)hv[e];
            }
        }
#pragma unroll
        for (int j = 0; j < 4; ++j) {
            v8h o8;
#pragma unroll
            for (int e = 0; e < 8; ++e) {
                int d = j * 8 + e;
                o8[e] = (_Float16)(coef[0] * kv[0][d] + coef[1] * kv[1][d] + coef[2] * kv[2][d]);
            }
            *(v8h*)(Mws + (size_t)n * C_ + cb + j * 8) = o8;
        }
    }
}

// ===========================================================================
// Kernel D: INT[b][hw][c] = attn_out_w x Mws + attn_out_b  (WMMA GEMM)
// ===========================================================================
__global__ __launch_bounds__(256) void outproj_kernel(const _Float16* __restrict__ Mws,
                                                      const _Float16* __restrict__ aowh,
                                                      const float* __restrict__ aob,
                                                      _Float16* __restrict__ INT) {
    __shared__ _Float16 XpT[TILE_N * 32];
    int n0 = blockIdx.x * TILE_N;
    int b = n0 >> 12, hw0 = n0 & (HW_ - 1);
    int tid = threadIdx.x, lane = tid & 31, m0 = (tid >> 5) * 32;
    int sp = tid & 63, k0 = (tid >> 6) * 8;
    v8f acc[2 * NT_];
#pragma unroll
    for (int i = 0; i < 2 * NT_; ++i) acc[i] = (v8f){};
    for (int ck = 0; ck < C_; ck += 32) {
        __syncthreads();
        *(v8h*)(XpT + sp * 32 + k0) = *(const v8h*)(Mws + (size_t)(n0 + sp) * C_ + ck + k0);
        __syncthreads();
        gemm_step_g(aowh, C_, ck, XpT, 32, 0, m0, lane, acc);
    }
#pragma unroll
    for (int mt = 0; mt < 2; ++mt)
#pragma unroll
        for (int nt = 0; nt < NT_; ++nt) {
            int p  = nt * 16 + (lane & 15);
            int rb = m0 + mt * 16 + ((lane >> 4) << 3);
            v8h hv;
#pragma unroll
            for (int v = 0; v < 8; ++v)
                hv[v] = (_Float16)(acc[mt * NT_ + nt][v] + aob[rb + v]);
            *(v8h*)(INT + (size_t)(b * HW_ + hw0 + p) * C_ + rb) = hv;
        }
}

// ===========================================================================
// Kernel E: 3x3 conv (implicit WMMA GEMM, 9 taps x 8 K-panels) + bias + BN +
// GELU, then 1x1 GEMM + bias + residual x[1] -> d_out (f32, NCHW)
// ===========================================================================
__global__ __launch_bounds__(256) void conv_kernel(const _Float16* __restrict__ INT,
                                                   const _Float16* __restrict__ w1h,
                                                   const float* __restrict__ b1,
                                                   const float* __restrict__ bn,
                                                   const _Float16* __restrict__ w2h,
                                                   const float* __restrict__ b2,
                                                   const float* __restrict__ x,
                                                   float* __restrict__ out) {
    __shared__ _Float16 XpT[TILE_N * 32];
    __shared__ _Float16 YpT[TILE_N * C_];
    int r0 = blockIdx.x;                  // image row
    int b  = blockIdx.y;
    int hw0 = r0 * 64;
    int tid = threadIdx.x, lane = tid & 31, m0 = (tid >> 5) * 32;
    int sp = tid & 63, k0 = (tid >> 6) * 8;

    v8f acc[2 * NT_];
#pragma unroll
    for (int i = 0; i < 2 * NT_; ++i) acc[i] = (v8f){};
    for (int tap = 0; tap < 9; ++tap) {
        int dy = tap / 3 - 1, dx = tap % 3 - 1;
        int hh = r0 + dy, ww = sp + dx;
        bool inb = (hh >= 0) && (hh < 64) && (ww >= 0) && (ww < 64);
        for (int ck = 0; ck < C_; ck += 32) {
            __syncthreads();
            v8h hv = (v8h){};
            if (inb)
                hv = *(const v8h*)(INT + (size_t)(b * HW_ + hh * 64 + ww) * C_ + ck + k0);
            *(v8h*)(XpT + sp * 32 + k0) = hv;
            __syncthreads();
            gemm_step_g(w1h + (size_t)tap * C_ * C_, C_, ck, XpT, 32, 0, m0, lane, acc);
        }
    }
    // epilogue: bias + BN + GELU -> YpT
#pragma unroll
    for (int mt = 0; mt < 2; ++mt)
#pragma unroll
        for (int nt = 0; nt < NT_; ++nt) {
            int p  = nt * 16 + (lane & 15);
            int rb = m0 + mt * 16 + ((lane >> 4) << 3);
            v8h hy;
#pragma unroll
            for (int v = 0; v < 8; ++v) {
                int o = rb + v;
                float y = acc[mt * NT_ + nt][v] + b1[o];
                float g = bn[o], be = bn[C_ + o], mu = bn[2 * C_ + o], va = bn[3 * C_ + o];
                y = (y - mu) * (g * rsqrtf(va + EPS_)) + be;
                hy[v] = (_Float16)gelu_exact(y);
            }
            *(v8h*)(YpT + p * C_ + rb) = hy;
        }
    __syncthreads();
    // 1x1 GEMM: global w2h x resident YpT
    v8f acc2[2 * NT_];
#pragma unroll
    for (int i = 0; i < 2 * NT_; ++i) acc2[i] = (v8f){};
#pragma unroll
    for (int ck = 0; ck < C_; ck += 32)
        gemm_step_g(w2h, C_, ck, YpT, C_, ck, m0, lane, acc2);
#pragma unroll
    for (int mt = 0; mt < 2; ++mt)
#pragma unroll
        for (int nt = 0; nt < NT_; ++nt) {
            int p  = nt * 16 + (lane & 15);
            int rb = m0 + mt * 16 + ((lane >> 4) << 3);
#pragma unroll
            for (int v = 0; v < 8; ++v) {
                int o = rb + v;
                float y = acc2[mt * NT_ + nt][v] + b2[o];
                y += x[((size_t)(1 * B_ + b) * C_ + o) * HW_ + hw0 + p];   // residual x[1]
                out[((size_t)(b * C_ + o)) * HW_ + hw0 + p] = y;
            }
        }
}

// ===========================================================================
// Workspace layout (all offsets 16B-aligned)
static const size_t OFF_POOLED = 0;                               // 12288 B
static const size_t OFF_WGT    = 12288;                           // 48 B
static const size_t OFF_WQH    = 16384;                           // 393216 B
static const size_t SZ_W       = (size_t)S_ * C_ * C_ * 2;        // 393216
static const size_t OFF_WKH    = OFF_WQH + SZ_W;
static const size_t OFF_WVH    = OFF_WKH + SZ_W;
static const size_t OFF_AIWH   = OFF_WVH + SZ_W;                  // 3C x C
static const size_t OFF_AOWH   = OFF_AIWH + SZ_W;                 // C x C
static const size_t OFF_W2H    = OFF_AOWH + (size_t)C_ * C_ * 2;
static const size_t OFF_W1H    = OFF_W2H + (size_t)C_ * C_ * 2;   // 9 x C x C
static const size_t OFF_QKV    = OFF_W1H + (size_t)9 * C_ * C_ * 2;
static const size_t OFF_MWS    = OFF_QKV + (size_t)9 * C_ * N_ * 2;
static const size_t OFF_INT    = OFF_MWS + (size_t)N_ * C_ * 2;

extern "C" void kernel_launch(void* const* d_in, const int* in_sizes, int n_in,
                              void* d_out, int out_size, void* d_ws, size_t ws_size,
                              hipStream_t stream) {
    const float* x    = (const float*)d_in[0];
    const float* Wq   = (const float*)d_in[1];
    const float* bq   = (const float*)d_in[2];
    const float* bnq  = (const float*)d_in[3];
    const float* Wk   = (const float*)d_in[4];
    const float* bk   = (const float*)d_in[5];
    const float* bnk  = (const float*)d_in[6];
    const float* Wv   = (const float*)d_in[7];
    const float* bv   = (const float*)d_in[8];
    const float* bnv  = (const float*)d_in[9];
    const float* iw1  = (const float*)d_in[10];
    const float* ib1  = (const float*)d_in[11];
    const float* iw2  = (const float*)d_in[12];
    const float* ib2  = (const float*)d_in[13];
    const float* aiw  = (const float*)d_in[14];
    const float* aib  = (const float*)d_in[15];
    const float* aow  = (const float*)d_in[16];
    const float* aob  = (const float*)d_in[17];
    const float* ow1  = (const float*)d_in[18];
    const float* ob1  = (const float*)d_in[19];
    const float* obn  = (const float*)d_in[20];
    const float* ow2  = (const float*)d_in[21];
    const float* ob2  = (const float*)d_in[22];

    char* ws = (char*)d_ws;
    float*    pooled = (float*)(ws + OFF_POOLED);
    float*    wgt    = (float*)(ws + OFF_WGT);
    _Float16* Wqh    = (_Float16*)(ws + OFF_WQH);
    _Float16* Wkh    = (_Float16*)(ws + OFF_WKH);
    _Float16* Wvh    = (_Float16*)(ws + OFF_WVH);
    _Float16* aiwh   = (_Float16*)(ws + OFF_AIWH);
    _Float16* aowh   = (_Float16*)(ws + OFF_AOWH);
    _Float16* w2h    = (_Float16*)(ws + OFF_W2H);
    _Float16* w1h    = (_Float16*)(ws + OFF_W1H);
    _Float16* qkv    = (_Float16*)(ws + OFF_QKV);
    _Float16* Mws    = (_Float16*)(ws + OFF_MWS);
    _Float16* INT    = (_Float16*)(ws + OFF_INT);
    float* out = (float*)d_out;

    // weight conversions (tiny, once)
    cvt_lin_kernel<<<(S_*C_*C_)/256, 256, 0, stream>>>(Wq,  Wqh,  S_*C_*C_);
    cvt_lin_kernel<<<(S_*C_*C_)/256, 256, 0, stream>>>(Wk,  Wkh,  S_*C_*C_);
    cvt_lin_kernel<<<(S_*C_*C_)/256, 256, 0, stream>>>(Wv,  Wvh,  S_*C_*C_);
    cvt_lin_kernel<<<(3*C_*C_)/256,  256, 0, stream>>>(aiw, aiwh, 3*C_*C_);
    cvt_lin_kernel<<<(C_*C_)/256,    256, 0, stream>>>(aow, aowh, C_*C_);
    cvt_lin_kernel<<<(C_*C_)/256,    256, 0, stream>>>(ow2, w2h,  C_*C_);
    cvt_conv_kernel<<<(9*C_*C_)/256, 256, 0, stream>>>(ow1, w1h);

    pool_kernel<<<S_ * B_ * C_, 256, 0, stream>>>(x, pooled);
    mlp_kernel<<<1, 256, 0, stream>>>(pooled, iw1, ib1, iw2, ib2, wgt);
    proj_kernel<<<dim3(N_ / TILE_N, S_), 256, 0, stream>>>(
        x, Wqh, Wkh, Wvh, bq, bnq, bk, bnk, bv, bnv, aiwh, aib, qkv);
    attn_small_kernel<<<N_ / 256, 256, 0, stream>>>(qkv, wgt, Mws);
    outproj_kernel<<<N_ / TILE_N, 256, 0, stream>>>(Mws, aowh, aob, INT);
    conv_kernel<<<dim3(64, B_), 256, 0, stream>>>(INT, w1h, ob1, obn, w2h, ob2, x, out);
}
